// MambaActionConditionedWorldModel_61203283968400
// MI455X (gfx1250) — matrix operand order repaired and checked by reading, not compiled, not run on hardware
//
#include <hip/hip_runtime.h>
#include <hip/hip_bf16.h>
#include <math.h>

// ---------------------------------------------------------------------------
// MI455X (gfx1250) implementation of MambaActionConditionedWorldModel.
// Compute-bound (~280 GFLOP of GEMM) -> v_wmma_f32_16x16x32_bf16, f32 accum.
// Double-buffered LDS pipeline, branch-free staging for full tiles.
// ---------------------------------------------------------------------------

typedef __bf16 bf16;
typedef bf16  bf16x8  __attribute__((ext_vector_type(8)));
typedef bf16  bf16x16 __attribute__((ext_vector_type(16)));
typedef float f32x8   __attribute__((ext_vector_type(8)));

#define LN_EPS 1e-5f
#define SHUF16(lo, hi) __builtin_shufflevector(lo, hi, 0,1,2,3,4,5,6,7,8,9,10,11,12,13,14,15)

constexpr int Bb = 64, Kk = 64, Pp = 8, Ee = 512, Aa = 64, Tt = 128;
constexpr int Ff   = Bb * Kk;   // 4096
constexpr int Rows = Ff * Pp;   // 32768
constexpr int Hh   = Ee * 2;    // 1024

enum { ACT_NONE = 0, ACT_SIG = 1, ACT_TANH = 2, ACT_GELU = 3 };

// ------------------------------ small helpers ------------------------------

__global__ void k_f32_to_bf16(const float* __restrict__ in, bf16* __restrict__ out, int n) {
  int i = blockIdx.x * blockDim.x + threadIdx.x;
  if (i < n) out[i] = (bf16)in[i];
}

// mean over T of ctx_tokens (B,T,E) -> bf16 (B,E)
__global__ void k_ctx_mean(const float* __restrict__ ctx, bf16* __restrict__ out) {
  int i = blockIdx.x * blockDim.x + threadIdx.x;  // b*E + e
  if (i >= Bb * Ee) return;
  int b = i / Ee, e = i % Ee;
  float s = 0.f;
  for (int t = 0; t < Tt; ++t) s += ctx[((size_t)b * Tt + t) * Ee + e];
  out[i] = (bf16)(s * (1.0f / Tt));
}

// W (Kd x Nd) f32 row-major  ->  Wt (Nd x Kd) bf16 row-major
__global__ void k_transpose_to_bf16(const float* __restrict__ in, bf16* __restrict__ out,
                                    int Kd, int Nd) {
  __shared__ float tile[32][33];
  int nx = blockIdx.x * 32 + threadIdx.x;
  int ky = blockIdx.y * 32 + threadIdx.y;
#pragma unroll
  for (int j = 0; j < 32; j += 8)
    if (ky + j < Kd && nx < Nd) tile[threadIdx.y + j][threadIdx.x] = in[(size_t)(ky + j) * Nd + nx];
  __syncthreads();
  int kx = blockIdx.y * 32 + threadIdx.x;
  int ny = blockIdx.x * 32 + threadIdx.y;
#pragma unroll
  for (int j = 0; j < 32; j += 8)
    if (ny + j < Nd && kx < Kd) out[(size_t)(ny + j) * Kd + kx] = (bf16)tile[threadIdx.x][threadIdx.y + j];
}

// block = 128 threads (4 waves, wave32). Reduces two values across the block.
__device__ inline void blk_red2(float& s0, float& s1, float* sh) {
#pragma unroll
  for (int off = 16; off > 0; off >>= 1) {
    s0 += __shfl_xor(s0, off);
    s1 += __shfl_xor(s1, off);
  }
  int w = threadIdx.x >> 5;
  if ((threadIdx.x & 31) == 0) { sh[w] = s0; sh[4 + w] = s1; }
  __syncthreads();
  s0 = sh[0] + sh[1] + sh[2] + sh[3];
  s1 = sh[4] + sh[5] + sh[6] + sh[7];
  __syncthreads();
}

// LayerNorm over rows of length E=512; 128 threads, 4 elems/thread.
template <typename Tout>
__global__ __launch_bounds__(128) void k_layernorm(const float* __restrict__ x,
                                                   const float* __restrict__ g,
                                                   const float* __restrict__ b,
                                                   Tout* __restrict__ y) {
  __shared__ float sh[8];
  const float* row = x + (size_t)blockIdx.x * Ee;
  float v[4], s0 = 0.f, s1 = 0.f;
#pragma unroll
  for (int i = 0; i < 4; ++i) {
    v[i] = row[threadIdx.x + i * 128];
    s0 += v[i];
    s1 += v[i] * v[i];
  }
  blk_red2(s0, s1, sh);
  float mean = s0 * (1.f / Ee);
  float rstd = rsqrtf(s1 * (1.f / Ee) - mean * mean + LN_EPS);
  Tout* yr = y + (size_t)blockIdx.x * Ee;
#pragma unroll
  for (int i = 0; i < 4; ++i) {
    int e = threadIdx.x + i * 128;
    yr[e] = (Tout)(((v[i] - mean) * rstd) * g[e] + b[e]);
  }
}

// seq[row] = LN(ap_lin[row]) + spB[b] + ctxB[b] + te[p] ;  row = f*P+p, b = f/K
__global__ __launch_bounds__(128) void k_ln_build_seq(const float* __restrict__ aplin,
                                                      const float* __restrict__ g,
                                                      const float* __restrict__ beta,
                                                      const float* __restrict__ spB,
                                                      const float* __restrict__ ctxB,
                                                      const float* __restrict__ te,
                                                      float* __restrict__ seq) {
  __shared__ float sh[8];
  int row = blockIdx.x;
  int p = row % Pp;
  int b = (row / Pp) / Kk;
  const float* xr = aplin + (size_t)row * Ee;
  float v[4], s0 = 0.f, s1 = 0.f;
#pragma unroll
  for (int i = 0; i < 4; ++i) {
    v[i] = xr[threadIdx.x + i * 128];
    s0 += v[i];
    s1 += v[i] * v[i];
  }
  blk_red2(s0, s1, sh);
  float mean = s0 * (1.f / Ee);
  float rstd = rsqrtf(s1 * (1.f / Ee) - mean * mean + LN_EPS);
#pragma unroll
  for (int i = 0; i < 4; ++i) {
    int e = threadIdx.x + i * 128;
    float ln = ((v[i] - mean) * rstd) * g[e] + beta[e];
    seq[(size_t)row * Ee + e] = ln + spB[(size_t)b * Ee + e] + ctxB[(size_t)b * Ee + e] + te[(size_t)p * Ee + e];
  }
}

// Mamba gate scan over P=8:  s = a*s + (1-a)*b ; seq += c*s
__global__ void k_scan(const float* __restrict__ a, const float* __restrict__ bg,
                       const float* __restrict__ c, float* __restrict__ seq) {
  int i = blockIdx.x * blockDim.x + threadIdx.x;  // f*E + e
  if (i >= Ff * Ee) return;
  int f = i / Ee, e = i % Ee;
  float s = 0.f;
  size_t base = (size_t)f * Pp * Ee + e;
#pragma unroll
  for (int t = 0; t < Pp; ++t) {
    size_t idx = base + (size_t)t * Ee;
    float at = a[idx], bt = bg[idx], ct = c[idx];
    s = at * s + (1.f - at) * bt;
    seq[idx] += ct * s;
  }
}

// trajectory scan: s = LN(s + delta_t); emits trajectory + terminal
__global__ __launch_bounds__(128) void k_traj(const float* __restrict__ z,
                                              const float* __restrict__ deltas,
                                              const float* __restrict__ sng,
                                              const float* __restrict__ snb,
                                              float* __restrict__ traj,
                                              float* __restrict__ term) {
  __shared__ float sh[8];
  int f = blockIdx.x;
  int b = f / Kk;
  float s[4];
#pragma unroll
  for (int i = 0; i < 4; ++i) s[i] = z[(size_t)b * Ee + threadIdx.x + i * 128];
  for (int t = 0; t < Pp; ++t) {
    float tv[4], s0 = 0.f, s1 = 0.f;
#pragma unroll
    for (int i = 0; i < 4; ++i) {
      tv[i] = s[i] + deltas[((size_t)f * Pp + t) * Ee + threadIdx.x + i * 128];
      s0 += tv[i];
      s1 += tv[i] * tv[i];
    }
    blk_red2(s0, s1, sh);
    float mean = s0 * (1.f / Ee);
    float rstd = rsqrtf(s1 * (1.f / Ee) - mean * mean + LN_EPS);
#pragma unroll
    for (int i = 0; i < 4; ++i) {
      int e = threadIdx.x + i * 128;
      s[i] = (tv[i] - mean) * rstd * sng[e] + snb[e];
      traj[((size_t)f * Pp + t) * Ee + e] = s[i];
    }
  }
#pragma unroll
  for (int i = 0; i < 4; ++i) {
    int e = threadIdx.x + i * 128;
    term[(size_t)f * Ee + e] = s[i];
  }
}

// summary = LN(mean over P of seq[f])
__global__ __launch_bounds__(128) void k_summary(const float* __restrict__ seq,
                                                 const float* __restrict__ g,
                                                 const float* __restrict__ b,
                                                 float* __restrict__ out) {
  __shared__ float sh[8];
  int f = blockIdx.x;
  float v[4], s0 = 0.f, s1 = 0.f;
#pragma unroll
  for (int i = 0; i < 4; ++i) {
    int e = threadIdx.x + i * 128;
    float m = 0.f;
    for (int t = 0; t < Pp; ++t) m += seq[((size_t)f * Pp + t) * Ee + e];
    v[i] = m * (1.f / Pp);
    s0 += v[i];
    s1 += v[i] * v[i];
  }
  blk_red2(s0, s1, sh);
  float mean = s0 * (1.f / Ee);
  float rstd = rsqrtf(s1 * (1.f / Ee) - mean * mean + LN_EPS);
#pragma unroll
  for (int i = 0; i < 4; ++i) {
    int e = threadIdx.x + i * 128;
    out[(size_t)f * Ee + e] = (v[i] - mean) * rstd * g[e] + b[e];
  }
}

// ------------------------------ WMMA GEMM ----------------------------------
// C[M,N] = act(A[M,K] @ Wt[N,K]^T + bias) (+ resid) ; A,Wt bf16, acc f32.
// Block tile 128x128, BK=32, 8 waves; wave = 2x4 grid of 16x16 WMMA tiles.
// Double-buffered LDS: global loads for tile k+1 issue before WMMAs of tile k.
// GUARD=false -> M is a multiple of 128, staging/stores are branch-free.
template <int ACT, bool RESID, bool GUARD, typename Tout>
__global__ __launch_bounds__(256) void k_gemm(const bf16* __restrict__ A,
                                              const bf16* __restrict__ Wt,
                                              const float* __restrict__ bias,
                                              const float* __restrict__ resid,
                                              Tout* __restrict__ out,
                                              int M, int N, int K) {
  constexpr int BM = 128, BN = 128, BK = 32, LDP = BK + 8;  // pad keeps 16B align
  __shared__ bf16 As[2][BM * LDP];
  __shared__ bf16 Bs[2][BN * LDP];

  const int tid  = threadIdx.x;
  const int lane = tid & 31;
  const int wave = tid >> 5;
  const int half = lane >> 4;   // K half-select per ISA wave32 layout
  const int l16  = lane & 15;
  const int m0 = blockIdx.y * BM;
  const int n0 = blockIdx.x * BN;
  const int wm = (wave >> 1) * 32;  // 4 waves along M
  const int wn = (wave & 1) * 64;   // 2 waves along N

  f32x8 acc[2][4];
#pragma unroll
  for (int i = 0; i < 2; ++i)
#pragma unroll
    for (int j = 0; j < 4; ++j)
#pragma unroll
      for (int r = 0; r < 8; ++r) acc[i][j][r] = 0.f;

  const int srow = tid >> 1;          // staging row 0..127
  const int scol = (tid & 1) * 16;    // 0 or 16 (elements)
  const bf16* aptr = A  + (size_t)(m0 + srow) * K + scol;
  const bf16* bptr = Wt + (size_t)(n0 + srow) * K + scol;

  auto stage = [&](int buf, int k0) {
    bf16x8 v0, v1;
    if (!GUARD || (m0 + srow) < M) {
      v0 = *(const bf16x8*)(aptr + k0);
      v1 = *(const bf16x8*)(aptr + k0 + 8);
    } else {
#pragma unroll
      for (int i = 0; i < 8; ++i) { v0[i] = (bf16)0.f; v1[i] = (bf16)0.f; }
    }
    bf16x8 w0 = *(const bf16x8*)(bptr + k0);
    bf16x8 w1 = *(const bf16x8*)(bptr + k0 + 8);
    *(bf16x8*)&As[buf][srow * LDP + scol]     = v0;
    *(bf16x8*)&As[buf][srow * LDP + scol + 8] = v1;
    *(bf16x8*)&Bs[buf][srow * LDP + scol]     = w0;
    *(bf16x8*)&Bs[buf][srow * LDP + scol + 8] = w1;
  };

  stage(0, 0);
  __syncthreads();

  const int nk = K / BK;
  for (int kb = 0; kb < nk; ++kb) {
    const int cur = kb & 1;
    if (kb + 1 < nk) stage(cur ^ 1, (kb + 1) * BK);  // overlap with WMMAs below

    // ---- fragments (ISA wave32 16-bit layouts) ----
    bf16x16 af[2];
#pragma unroll
    for (int mt = 0; mt < 2; ++mt) {
      int r  = wm + mt * 16 + l16;
      int kb2 = half * 8;  // A: VGPR0-3 -> K=kb2..kb2+7, VGPR4-7 -> K=kb2+16..kb2+23
      bf16x8 lo = *(const bf16x8*)&As[cur][r * LDP + kb2];
      bf16x8 hi = *(const bf16x8*)&As[cur][r * LDP + kb2 + 16];
      af[mt] = SHUF16(lo, hi);
    }
    bf16x16 bfr[4];
#pragma unroll
    for (int nt = 0; nt < 4; ++nt) {
      int r  = wn + nt * 16 + l16;
      int kb2 = half * 8;  // B: VGPR0-3 -> K=kb2..kb2+7, VGPR4-7 -> K=kb2+16..kb2+23
      bf16x8 lo = *(const bf16x8*)&Bs[cur][r * LDP + kb2];
      bf16x8 hi = *(const bf16x8*)&Bs[cur][r * LDP + kb2 + 16];
      bfr[nt] = SHUF16(lo, hi);
    }
#pragma unroll
    for (int mt = 0; mt < 2; ++mt)
#pragma unroll
      for (int nt = 0; nt < 4; ++nt)
        acc[mt][nt] = __builtin_amdgcn_wmma_f32_16x16x32_bf16(
            false, af[mt], false, bfr[nt], (short)0, acc[mt][nt], false, false);
    __syncthreads();
  }

  // ---- epilogue: bias + activation (+ residual), store f32/bf16 ----
#pragma unroll
  for (int mt = 0; mt < 2; ++mt) {
#pragma unroll
    for (int nt = 0; nt < 4; ++nt) {
      int col = n0 + wn + nt * 16 + l16;
      float bval = bias[col];
#pragma unroll
      for (int r = 0; r < 8; ++r) {
        int row = m0 + wm + mt * 16 + half * 8 + r;
        if (!GUARD || row < M) {
          float x = acc[mt][nt][r] + bval;
          if constexpr (ACT == ACT_SIG)  x = 1.f / (1.f + expf(-x));
          if constexpr (ACT == ACT_TANH) x = tanhf(x);
          if constexpr (ACT == ACT_GELU) x = 0.5f * x * (1.f + erff(x * 0.70710678118654752f));
          size_t oi = (size_t)row * N + col;
          if constexpr (RESID) x += resid[oi];
          out[oi] = (Tout)x;
        }
      }
    }
  }
}

// ------------------------------ host driver --------------------------------

extern "C" void kernel_launch(void* const* d_in, const int* in_sizes, int n_in,
                              void* d_out, int out_size, void* d_ws, size_t ws_size,
                              hipStream_t stream) {
  (void)in_sizes; (void)n_in; (void)out_size; (void)ws_size;

  // Top-level dict order: z, actions, ctx_tokens, params (pytree: sorted keys).
  const float* z       = (const float*)d_in[0];
  const float* actions = (const float*)d_in[1];
  const float* ctx     = (const float*)d_in[2];
  const float* ap_W    = (const float*)d_in[3];
  const float* ap_b    = (const float*)d_in[4];
  const float* ap_beta = (const float*)d_in[5];
  const float* ap_g    = (const float*)d_in[6];
  const float* cx_W    = (const float*)d_in[7];
  const float* cx_b    = (const float*)d_in[8];
  const float* cx_beta = (const float*)d_in[9];
  const float* cx_g    = (const float*)d_in[10];
  const float *L_W1[2], *L_W2[2], *L_Wi[2], *L_Wo[2], *L_Ws[2], *L_b1[2], *L_b2[2],
      *L_bi[2], *L_bo[2], *L_bs[2], *L_n1b[2], *L_n1g[2], *L_n2b[2], *L_n2g[2];
  for (int l = 0; l < 2; ++l) {
    int o = 11 + 14 * l;
    L_W1[l]  = (const float*)d_in[o + 0];
    L_W2[l]  = (const float*)d_in[o + 1];
    L_Wi[l]  = (const float*)d_in[o + 2];
    L_Wo[l]  = (const float*)d_in[o + 3];
    L_Ws[l]  = (const float*)d_in[o + 4];
    L_b1[l]  = (const float*)d_in[o + 5];
    L_b2[l]  = (const float*)d_in[o + 6];
    L_bi[l]  = (const float*)d_in[o + 7];
    L_bo[l]  = (const float*)d_in[o + 8];
    L_bs[l]  = (const float*)d_in[o + 9];
    L_n1b[l] = (const float*)d_in[o + 10];
    L_n1g[l] = (const float*)d_in[o + 11];
    L_n2b[l] = (const float*)d_in[o + 12];
    L_n2g[l] = (const float*)d_in[o + 13];
  }
  const float* sn_b    = (const float*)d_in[39];
  const float* sn_g    = (const float*)d_in[40];
  const float* st_W    = (const float*)d_in[41];
  const float* st_b    = (const float*)d_in[42];
  const float* st_beta = (const float*)d_in[43];
  const float* st_g    = (const float*)d_in[44];
  const float* sum_b   = (const float*)d_in[45];
  const float* sum_g   = (const float*)d_in[46];
  const float* th_W1   = (const float*)d_in[47];
  const float* th_b1   = (const float*)d_in[48];
  const float* th_W2   = (const float*)d_in[49];
  const float* th_b2   = (const float*)d_in[50];
  const float* te      = (const float*)d_in[51];

  float* out_traj = (float*)d_out;                          // (F,P,E)
  float* out_term = out_traj + (size_t)Ff * Pp * Ee;        // (F,E)
  float* out_sum  = out_term + (size_t)Ff * Ee;             // (F,E)

  // deterministic bump allocator over d_ws
  char* wsp = (char*)d_ws;
  size_t off = 0;
  auto alloc = [&](size_t bytes) -> void* {
    void* p = wsp + off;
    off += (bytes + 255) & ~(size_t)255;
    return p;
  };
  bf16* wt_st = (bf16*)alloc((size_t)Ee * Ee * 2);
  bf16* wt_cx = (bf16*)alloc((size_t)Ee * Ee * 2);
  bf16* wt_ap = (bf16*)alloc((size_t)Ee * Aa * 2);
  bf16 *wt_s[2], *wt_i[2], *wt_o[2], *wt_1[2], *wt_2[2];
  for (int l = 0; l < 2; ++l) {
    wt_s[l] = (bf16*)alloc((size_t)Ee * Ee * 2);
    wt_i[l] = (bf16*)alloc((size_t)Ee * Ee * 2);
    wt_o[l] = (bf16*)alloc((size_t)Ee * Ee * 2);
    wt_1[l] = (bf16*)alloc((size_t)Hh * Ee * 2);
    wt_2[l] = (bf16*)alloc((size_t)Ee * Hh * 2);
  }
  bf16* wt_t1 = (bf16*)alloc((size_t)Ee * Ee * 2);
  bf16* wt_t2 = (bf16*)alloc((size_t)Ee * Ee * 2);
  bf16*  z_bf   = (bf16*)alloc((size_t)Bb * Ee * 2);
  bf16*  cm_bf  = (bf16*)alloc((size_t)Bb * Ee * 2);
  bf16*  act_bf = (bf16*)alloc((size_t)Rows * Aa * 2);
  float* spB    = (float*)alloc((size_t)Bb * Ee * 4);
  float* cxB    = (float*)alloc((size_t)Bb * Ee * 4);
  float* tmp64  = (float*)alloc((size_t)Bb * Ee * 4);
  float* seq    = (float*)alloc((size_t)Rows * Ee * 4);
  bf16*  xb     = (bf16*)alloc((size_t)Rows * Ee * 2);
  float* bufA   = (float*)alloc((size_t)Rows * Ee * 4);  // also h  (Rows x Hh bf16)
  float* bufB   = (float*)alloc((size_t)Rows * Ee * 4);  // also deltas
  float* bufC   = (float*)alloc((size_t)Rows * Ee * 4);  // also h2 (Rows x Ee bf16)

  auto tr = [&](const float* W, bf16* Wt, int Kd, int Nd) {
    dim3 g((Nd + 31) / 32, (Kd + 31) / 32), b(32, 8);
    k_transpose_to_bf16<<<g, b, 0, stream>>>(W, Wt, Kd, Nd);
  };
  auto ggrid = [](int M, int N) { return dim3((unsigned)(N / 128), (unsigned)((M + 127) / 128)); };

  // ---- conversions / weight prep ----
  k_f32_to_bf16<<<(Bb * Ee + 255) / 256, 256, 0, stream>>>(z, z_bf, Bb * Ee);
  k_f32_to_bf16<<<(Rows * Aa + 255) / 256, 256, 0, stream>>>(actions, act_bf, Rows * Aa);
  k_ctx_mean<<<(Bb * Ee + 255) / 256, 256, 0, stream>>>(ctx, cm_bf);
  tr(st_W, wt_st, Ee, Ee);
  tr(cx_W, wt_cx, Ee, Ee);
  tr(ap_W, wt_ap, Aa, Ee);
  for (int l = 0; l < 2; ++l) {
    tr(L_Ws[l], wt_s[l], Ee, Ee);
    tr(L_Wi[l], wt_i[l], Ee, Ee);
    tr(L_Wo[l], wt_o[l], Ee, Ee);
    tr(L_W1[l], wt_1[l], Ee, Hh);
    tr(L_W2[l], wt_2[l], Hh, Ee);
  }
  tr(th_W1, wt_t1, Ee, Ee);
  tr(th_W2, wt_t2, Ee, Ee);

  // ---- projections (sp/ctx only need the B=64 distinct rows; M=64 -> GUARD) --
  k_gemm<ACT_NONE, false, true,  float><<<ggrid(Bb, Ee), 256, 0, stream>>>(z_bf, wt_st, st_b, nullptr, tmp64, Bb, Ee, Ee);
  k_layernorm<float><<<Bb, 128, 0, stream>>>(tmp64, st_g, st_beta, spB);
  k_gemm<ACT_NONE, false, true,  float><<<ggrid(Bb, Ee), 256, 0, stream>>>(cm_bf, wt_cx, cx_b, nullptr, tmp64, Bb, Ee, Ee);
  k_layernorm<float><<<Bb, 128, 0, stream>>>(tmp64, cx_g, cx_beta, cxB);
  k_gemm<ACT_NONE, false, false, float><<<ggrid(Rows, Ee), 256, 0, stream>>>(act_bf, wt_ap, ap_b, nullptr, bufA, Rows, Ee, Aa);
  k_ln_build_seq<<<Rows, 128, 0, stream>>>(bufA, ap_g, ap_beta, spB, cxB, te, seq);

  // ---- transformer/mamba layers ----
  for (int l = 0; l < 2; ++l) {
    k_layernorm<bf16><<<Rows, 128, 0, stream>>>(seq, L_n1g[l], L_n1b[l], xb);
    k_gemm<ACT_SIG,  false, false, float><<<ggrid(Rows, Ee), 256, 0, stream>>>(xb, wt_s[l], L_bs[l], nullptr, bufA, Rows, Ee, Ee);
    k_gemm<ACT_TANH, false, false, float><<<ggrid(Rows, Ee), 256, 0, stream>>>(xb, wt_i[l], L_bi[l], nullptr, bufB, Rows, Ee, Ee);
    k_gemm<ACT_SIG,  false, false, float><<<ggrid(Rows, Ee), 256, 0, stream>>>(xb, wt_o[l], L_bo[l], nullptr, bufC, Rows, Ee, Ee);
    k_scan<<<(Ff * Ee + 255) / 256, 256, 0, stream>>>(bufA, bufB, bufC, seq);
    k_layernorm<bf16><<<Rows, 128, 0, stream>>>(seq, L_n2g[l], L_n2b[l], xb);
    k_gemm<ACT_GELU, false, false, bf16><<<ggrid(Rows, Hh), 256, 0, stream>>>(xb, wt_1[l], L_b1[l], nullptr, (bf16*)bufA, Rows, Hh, Ee);
    k_gemm<ACT_NONE, true,  false, float><<<ggrid(Rows, Ee), 256, 0, stream>>>((bf16*)bufA, wt_2[l], L_b2[l], seq, seq, Rows, Ee, Hh);
  }

  // ---- trajectory head ----
  k_layernorm<bf16><<<Rows, 128, 0, stream>>>(seq, sum_g, sum_b, xb);  // normed
  k_gemm<ACT_GELU, false, false, bf16><<<ggrid(Rows, Ee), 256, 0, stream>>>(xb, wt_t1, th_b1, nullptr, (bf16*)bufC, Rows, Ee, Ee);
  k_gemm<ACT_NONE, false, false, float><<<ggrid(Rows, Ee), 256, 0, stream>>>((bf16*)bufC, wt_t2, th_b2, nullptr, bufB, Rows, Ee, Ee);
  k_traj<<<Ff, 128, 0, stream>>>(z, bufB, sn_g, sn_b, out_traj, out_term);
  k_summary<<<Ff, 128, 0, stream>>>(seq, sum_g, sum_b, out_sum);
}
